// Policy_RNN_60069412601897
// MI455X (gfx1250) — matrix operand order repaired and checked by reading, not compiled, not run on hardware
//
#include <hip/hip_runtime.h>
#include <hip/hip_bf16.h>

typedef __attribute__((ext_vector_type(16))) _Float16 v16h;
typedef __attribute__((ext_vector_type(8)))  float    v8f;

#define BN_EPS 1e-5f

// ---- optional CDNA5 async global->LDS path (guarded; falls back cleanly) ----
#if defined(__AMDGCN__) && __has_builtin(__builtin_amdgcn_global_load_async_to_lds_b32)
#define USE_ASYNC_LDS 1
__device__ __forceinline__ void g2lds_f32(const float* g, float* l) {
  // builtin signature (from hipcc diagnostic): (int AS1*, int AS3*, imm offset, imm cpol)
  __builtin_amdgcn_global_load_async_to_lds_b32(
      (__attribute__((address_space(1))) int*)g,
      (__attribute__((address_space(3))) int*)l, /*offset=*/0, /*cpol=*/0);
}
__device__ __forceinline__ void wait_async() {
#if __has_builtin(__builtin_amdgcn_s_wait_asynccnt)
  __builtin_amdgcn_s_wait_asynccnt(0);
#else
  asm volatile("s_wait_asynccnt 0" ::: "memory");
#endif
}
#endif

// ---------------------------------------------------------------------------
// Zero the BN-statistics workspace (8 stages x [sum, sumsq] x <=64 channels).
// ---------------------------------------------------------------------------
__global__ __launch_bounds__(256) void k_init_stats(float* __restrict__ stats) {
  int i = blockIdx.x * 256 + threadIdx.x;
  if (i < 1024) stats[i] = 0.0f;
}

// ---------------------------------------------------------------------------
// Depthwise 3x3 stride-2 pad-1 conv.
//  - Optionally applies producer BN + ReLU to the input (statsIn != null).
//  - Writes raw conv output and accumulates per-channel sum/sumsq for the
//    *next* BN via LDS reduction + one global atomic per channel per block.
// ---------------------------------------------------------------------------
__global__ __launch_bounds__(256) void k_dwconv(
    const float* __restrict__ in, float* __restrict__ out,
    const float* __restrict__ w,                     // C*9 (OIHW depthwise)
    const float* __restrict__ gIn, const float* __restrict__ bIn,
    const float* __restrict__ statsIn,               // [2*C] of input, or null
    float invCntIn,
    float* __restrict__ statsOut,                    // [2*C] of output
    int C, int Hin, int Win, int Hout, int Wout, int total)
{
  __shared__ float sred[128];                        // up to 64 ch * {sum,sq}
  int tid = threadIdx.x;
  if (tid < 2 * C) sred[tid] = 0.0f;
  __syncthreads();

  int idx = blockIdx.x * 256 + tid;
  if (idx < total) {
    int wo = idx % Wout;
    int t1 = idx / Wout;
    int ho = t1 % Hout;
    int t2 = t1 / Hout;
    int c  = t2 % C;
    int n  = t2 / C;

    float s = 1.0f, sh = 0.0f;
    const bool act = (statsIn != nullptr);
    if (act) {                                       // fold BN of producer
      float m = statsIn[c] * invCntIn;
      float v = statsIn[C + c] * invCntIn - m * m;
      s  = gIn[c] * rsqrtf(v + BN_EPS);
      sh = bIn[c] - m * s;
    }

    const float* wc   = w + c * 9;
    const float* base = in + ((long)n * C + c) * Hin * Win;
    float acc = 0.0f;
    #pragma unroll
    for (int kh = 0; kh < 3; ++kh) {
      int hi = ho * 2 - 1 + kh;
      if (hi < 0 || hi >= Hin) continue;
      #pragma unroll
      for (int kw = 0; kw < 3; ++kw) {
        int wi = wo * 2 - 1 + kw;
        if (wi < 0 || wi >= Win) continue;
        float v = base[hi * Win + wi];
        if (act) v = fmaxf(fmaf(v, s, sh), 0.0f);    // BN + ReLU of input
        acc += v * wc[kh * 3 + kw];
      }
    }
    out[idx] = acc;
    atomicAdd(&sred[c],     acc);
    atomicAdd(&sred[C + c], acc * acc);
  }
  __syncthreads();
  if (tid < 2 * C) atomicAdd(&statsOut[tid], sred[tid]);
}

// ---------------------------------------------------------------------------
// Pointwise (1x1) conv as WMMA GEMM:  D[pixel, outCh] = A[pixel, k] * B[k, outCh]
//   All of CIN/COUT/NCHUNKS/HW are compile-time:
//    - padding predicates fold away (CIN=64) or become v_cndmask (CIN=10)
//    - pixel -> (n, hw) decomposition divides by constant HW, so it strength-
//      reduces to mul_hi+shift (no software u64 division sequences)
//   Loads are unconditional -> batched in s_clause groups under one wait.
// A-fragment lane mapping per ISA 7.12.2 (16-bit A 16x32):
//   lane L: row M = L&15; kb = (L<16)?0:8; elem e -> K = e<8 ? kb+e : 16+kb+e-8
// D mapping: lane L: N = L&15; acc[r] -> M = r + 8*(L>=16)
// ---------------------------------------------------------------------------
template <int CIN, int COUT, int NCHUNKS, int HW>
__global__ __launch_bounds__(256) void k_pw_wmma(
    const float* __restrict__ in,                    // raw dw out (N,CIN,HW)
    const float* __restrict__ statsIn,               // [2*CIN]
    const float* __restrict__ gIn, const float* __restrict__ bIn,
    float invCntIn,
    const float* __restrict__ w,                     // COUT x CIN
    float* __restrict__ out,                         // (N,COUT,HW)
    float* __restrict__ statsOut)                    // [2*COUT]
{
  __shared__ float2 sSS[CIN];                        // {scale, shift} per ch
  __shared__ float  sred[32];
  int tid = threadIdx.x;
  if (tid < 32) sred[tid] = 0.0f;
  if (tid < CIN) {
    float m = statsIn[tid] * invCntIn;
    float v = statsIn[CIN + tid] * invCntIn - m * m;
    float s = gIn[tid] * rsqrtf(v + BN_EPS);
    sSS[tid] = make_float2(s, bIn[tid] - m * s);
  }
  __syncthreads();

  int lane    = tid & 31;
  int waveId  = tid >> 5;
  int mRow    = lane & 15;
  int hiHalf  = lane >> 4;
  unsigned pixelBase = ((unsigned)blockIdx.x * 8 + waveId) * 16;

  // Pixel this lane supplies for the A fragment (div by constant HW)
  unsigned pA  = pixelBase + mRow;
  unsigned nA  = pA / (unsigned)HW;
  unsigned hwA = pA % (unsigned)HW;
  const float* inRow = in + (size_t)nA * CIN * HW + hwA;
  int kb = hiHalf ? 8 : 0;

  v8f acc = {};
  #pragma unroll
  for (int t = 0; t < NCHUNKS; ++t) {
    v16h a, b;
    // ---- gather phase: unconditional clamped loads (batch under one wait)
    float  rawv[16];
    float2 ss[16];
    int    kk[16];
    #pragma unroll
    for (int e = 0; e < 16; ++e) {
      int kl = (e < 8) ? (kb + e) : (16 + kb + (e - 8));
      int k  = t * 32 + kl;
      int kc = (k < CIN) ? k : (CIN - 1);             // folds away for CIN=64
      kk[e]   = k;
      rawv[e] = inRow[(size_t)kc * HW];
      ss[e]   = sSS[kc];
    }
    // ---- transform phase: BN + ReLU + f16 convert, zero padded lanes
    #pragma unroll
    for (int e = 0; e < 16; ++e) {
      float va = fmaxf(fmaf(rawv[e], ss[e].x, ss[e].y), 0.0f);
      if (kk[e] >= CIN) va = 0.0f;                    // v_cndmask, no branch
      a[e] = (_Float16)va;
    }
    // ---- B fragment: lane&15 = N col, e = K (same order as A)
    int  nB  = lane & 15;
    #pragma unroll
    for (int e = 0; e < 16; ++e) {
      int  kB  = t * 32 + ((e < 8) ? (kb + e) : (16 + kb + (e - 8)));
      int  kBc = (kB < CIN) ? kB : (CIN - 1);
      bool kBv = (kB < CIN);
      float vb = 0.0f;
      if (nB < COUT) {
        vb = w[nB * CIN + kBc];
        if (!kBv) vb = 0.0f;
      }
      b[e] = (_Float16)vb;
    }
    acc = __builtin_amdgcn_wmma_f32_16x16x32_f16(
        /*neg_a=*/false, a, /*neg_b=*/false, b,
        /*c_mod=*/(short)0, acc, /*reuse_a=*/false, /*reuse_b=*/false);
  }

  int ch = lane & 15;
  #pragma unroll
  for (int r = 0; r < 8; ++r) {
    unsigned p  = pixelBase + r + 8 * hiHalf;
    unsigned n  = p / (unsigned)HW;                   // mul_hi+shift (const HW)
    unsigned hw = p % (unsigned)HW;
    float v = acc[r];
    if (ch < COUT) {
      out[((size_t)n * COUT + ch) * HW + hw] = v;
      atomicAdd(&sred[ch],      v);
      atomicAdd(&sred[16 + ch], v * v);
    }
  }
  __syncthreads();
  if (tid < COUT) {
    atomicAdd(&statsOut[tid],        sred[tid]);
    atomicAdd(&statsOut[COUT + tid], sred[16 + tid]);
  }
}

// ---------------------------------------------------------------------------
// Final BN + ReLU + 4x4 global average pool -> feat (256,10)
// ---------------------------------------------------------------------------
__global__ __launch_bounds__(256) void k_pool(
    const float* __restrict__ in, const float* __restrict__ stats,
    const float* __restrict__ g, const float* __restrict__ b,
    float invCnt, float* __restrict__ feat)
{
  int idx = blockIdx.x * 256 + threadIdx.x;
  if (idx >= 2560) return;
  int n = idx / 10, c = idx % 10;
  float m = stats[c] * invCnt;
  float v = stats[10 + c] * invCnt - m * m;
  float s  = g[c] * rsqrtf(v + BN_EPS);
  float sh = b[c] - m * s;
  const float* p = in + ((long)n * 10 + c) * 16;
  float acc = 0.0f;
  #pragma unroll
  for (int j = 0; j < 16; ++j) acc += fmaxf(fmaf(p[j], s, sh), 0.0f);
  feat[idx] = acc * (1.0f / 16.0f);
}

// ---------------------------------------------------------------------------
// LSTM scan (19 steps, 2 cells, head, categorical sampling). One block of 256
// threads; thread n = batch element n. Weights staged into LDS with async
// global->LDS copies (ASYNCcnt path) when the toolchain provides the builtin.
// ---------------------------------------------------------------------------
__device__ __forceinline__ float sigf(float x) { return 1.0f / (1.0f + __expf(-x)); }

__device__ __forceinline__ float urand(unsigned h) {   // murmur3 finalizer
  h ^= h >> 16; h *= 0x85ebca6bu; h ^= h >> 13; h *= 0xc2b2ae35u; h ^= h >> 16;
  return (float)(h >> 8) * (1.0f / 16777216.0f);
}

__device__ __forceinline__ void sample6(const float* l, unsigned seed,
                                        int& pick, float& lp) {
  float mx = l[0];
  #pragma unroll
  for (int i = 1; i < 6; ++i) mx = fmaxf(mx, l[i]);
  float e[6], Z = 0.0f;
  #pragma unroll
  for (int i = 0; i < 6; ++i) { e[i] = __expf(l[i] - mx); Z += e[i]; }
  float target = urand(seed) * Z, cum = 0.0f, lsel = l[5];
  pick = 5;
  bool done = false;
  #pragma unroll
  for (int i = 0; i < 6; ++i) {
    cum += e[i];
    if (!done && target <= cum) { pick = i; lsel = l[i]; done = true; }
  }
  lp = lsel - mx - __logf(Z);
}

__global__ __launch_bounds__(256) void k_lstm(
    const float* __restrict__ feat,
    const float* __restrict__ wih0, const float* __restrict__ whh0,
    const float* __restrict__ bih0, const float* __restrict__ bhh0,
    const float* __restrict__ wih1, const float* __restrict__ whh1,
    const float* __restrict__ bih1, const float* __restrict__ bhh1,
    const float* __restrict__ head_w, const float* __restrict__ head_b,
    float* __restrict__ outp)
{
  __shared__ float Lwih0[480], Lwhh0[400], Lwih1[400], Lwhh1[400],
                   Lhw[120], Lhb[12], Lb0[40], Lb1[40], Ltb[160];
  int tid = threadIdx.x;

#ifdef USE_ASYNC_LDS
  // contiguous global->LDS copies on the ASYNCcnt path
  for (int i = tid; i < 480; i += 256) g2lds_f32(wih0 + i,   &Lwih0[i]);
  for (int i = tid; i < 400; i += 256) {
    g2lds_f32(whh0 + i, &Lwhh0[i]);
    g2lds_f32(wih1 + i, &Lwih1[i]);
    g2lds_f32(whh1 + i, &Lwhh1[i]);
  }
  for (int i = tid; i < 120; i += 256) g2lds_f32(head_w + i, &Lhw[i]);
  if (tid < 12) g2lds_f32(head_b + tid, &Lhb[tid]);
  if (tid < 40) {
    g2lds_f32(bih0 + tid, &Ltb[tid]);
    g2lds_f32(bhh0 + tid, &Ltb[40 + tid]);
    g2lds_f32(bih1 + tid, &Ltb[80 + tid]);
    g2lds_f32(bhh1 + tid, &Ltb[120 + tid]);
  }
  wait_async();
  __syncthreads();
  if (tid < 40) {
    Lb0[tid] = Ltb[tid] + Ltb[40 + tid];
    Lb1[tid] = Ltb[80 + tid] + Ltb[120 + tid];
  }
#else
  for (int i = tid; i < 480; i += 256) Lwih0[i] = wih0[i];
  for (int i = tid; i < 400; i += 256) { Lwhh0[i] = whh0[i]; Lwih1[i] = wih1[i]; Lwhh1[i] = whh1[i]; }
  for (int i = tid; i < 120; i += 256) Lhw[i] = head_w[i];
  if (tid < 12) Lhb[tid] = head_b[tid];
  if (tid < 40) { Lb0[tid] = bih0[tid] + bhh0[tid]; Lb1[tid] = bih1[tid] + bhh1[tid]; }
#endif
  __syncthreads();

  const int n = tid;
  float f[10], h0[10], c0[10], h1[10], c1[10];
  #pragma unroll
  for (int j = 0; j < 10; ++j) {
    f[j] = feat[n * 10 + j];
    h0[j] = c0[j] = h1[j] = c1[j] = 0.0f;
  }
  float pa = 0.0f, pw = 0.0f;

  for (int t = 0; t < 19; ++t) {
    float inp[12];
    #pragma unroll
    for (int j = 0; j < 10; ++j) inp[j] = f[j];
    inp[10] = pa; inp[11] = pw;

    // ---- cell 0 (input 12) ----
    float nh0[10];
    #pragma unroll
    for (int j = 0; j < 10; ++j) {
      float gi = Lb0[j], gf = Lb0[10 + j], gg = Lb0[20 + j], go = Lb0[30 + j];
      #pragma unroll
      for (int k = 0; k < 12; ++k) {
        float x = inp[k];
        gi = fmaf(Lwih0[j * 12 + k], x, gi);
        gf = fmaf(Lwih0[(10 + j) * 12 + k], x, gf);
        gg = fmaf(Lwih0[(20 + j) * 12 + k], x, gg);
        go = fmaf(Lwih0[(30 + j) * 12 + k], x, go);
      }
      #pragma unroll
      for (int k = 0; k < 10; ++k) {
        float x = h0[k];
        gi = fmaf(Lwhh0[j * 10 + k], x, gi);
        gf = fmaf(Lwhh0[(10 + j) * 10 + k], x, gf);
        gg = fmaf(Lwhh0[(20 + j) * 10 + k], x, gg);
        go = fmaf(Lwhh0[(30 + j) * 10 + k], x, go);
      }
      float cc = sigf(gf) * c0[j] + sigf(gi) * tanhf(gg);
      c0[j] = cc;
      nh0[j] = sigf(go) * tanhf(cc);
    }
    #pragma unroll
    for (int j = 0; j < 10; ++j) h0[j] = nh0[j];

    // ---- cell 1 (input 10) ----
    float nh1[10];
    #pragma unroll
    for (int j = 0; j < 10; ++j) {
      float gi = Lb1[j], gf = Lb1[10 + j], gg = Lb1[20 + j], go = Lb1[30 + j];
      #pragma unroll
      for (int k = 0; k < 10; ++k) {
        float x = h0[k];
        gi = fmaf(Lwih1[j * 10 + k], x, gi);
        gf = fmaf(Lwih1[(10 + j) * 10 + k], x, gf);
        gg = fmaf(Lwih1[(20 + j) * 10 + k], x, gg);
        go = fmaf(Lwih1[(30 + j) * 10 + k], x, go);
      }
      #pragma unroll
      for (int k = 0; k < 10; ++k) {
        float x = h1[k];
        gi = fmaf(Lwhh1[j * 10 + k], x, gi);
        gf = fmaf(Lwhh1[(10 + j) * 10 + k], x, gf);
        gg = fmaf(Lwhh1[(20 + j) * 10 + k], x, gg);
        go = fmaf(Lwhh1[(30 + j) * 10 + k], x, go);
      }
      float cc = sigf(gf) * c1[j] + sigf(gi) * tanhf(gg);
      c1[j] = cc;
      nh1[j] = sigf(go) * tanhf(cc);
    }
    #pragma unroll
    for (int j = 0; j < 10; ++j) h1[j] = nh1[j];

    // ---- head + sampling ----
    float logits[12];
    #pragma unroll
    for (int e = 0; e < 12; ++e) {
      float acc = Lhb[e];
      #pragma unroll
      for (int j = 0; j < 10; ++j) acc = fmaf(Lhw[e * 10 + j], h1[j], acc);
      logits[e] = acc;
    }
    unsigned base = ((unsigned)(n * 19 + t)) * 0x9E3779B9u;
    int ba, bw; float lpa, lpw;
    sample6(logits,     base ^ 0x1234567u, ba, lpa);
    sample6(logits + 6, base ^ 0x89ABCDEu, bw, lpw);

    int o = n * 19 + t;
    outp[o]         = (float)(ba + 2);
    outp[4864  + o] = (float)(bw + 2);
    outp[9728  + o] = lpa;
    outp[14592 + o] = lpw;
    pa = (float)ba;
    pw = (float)bw;
  }
}

// ---------------------------------------------------------------------------
// Orchestration
// ---------------------------------------------------------------------------
extern "C" void kernel_launch(void* const* d_in, const int* in_sizes, int n_in,
                              void* d_out, int out_size, void* d_ws, size_t ws_size,
                              hipStream_t stream) {
  const float* x      = (const float*)d_in[0];
  const float* dw0_w  = (const float*)d_in[1];
  const float* bn0a_g = (const float*)d_in[2];
  const float* bn0a_b = (const float*)d_in[3];
  const float* pw0_w  = (const float*)d_in[4];
  const float* bn0b_g = (const float*)d_in[5];
  const float* bn0b_b = (const float*)d_in[6];
  const float* dwr_w  = (const float*)d_in[7];   // (3,10,1,3,3)
  const float* bnra_g = (const float*)d_in[8];   // (3,10)
  const float* bnra_b = (const float*)d_in[9];
  const float* pwr_w  = (const float*)d_in[10];  // (3,10,10,1,1)
  const float* bnrb_g = (const float*)d_in[11];
  const float* bnrb_b = (const float*)d_in[12];
  const float* wih0   = (const float*)d_in[13];
  const float* whh0   = (const float*)d_in[14];
  const float* bih0   = (const float*)d_in[15];
  const float* bhh0   = (const float*)d_in[16];
  const float* wih1   = (const float*)d_in[17];
  const float* whh1   = (const float*)d_in[18];
  const float* bih1   = (const float*)d_in[19];
  const float* bhh1   = (const float*)d_in[20];
  const float* head_w = (const float*)d_in[21];
  const float* head_b = (const float*)d_in[22];
  float* outp = (float*)d_out;

  // workspace layout (floats):
  //   [0,1024)    stats: 8 slots of 128 (dw0,pw0,dw1,pw1,dw2,pw2,dw3,pw3)
  //   [1024,3584) feat (256*10)
  //   [4096, +12845056)  bufA (max dw output: 256*64*28*28)
  //   bufB = bufA + 12845056 (max pw output: 256*10*28*28 = 2007040)
  float* ws    = (float*)d_ws;
  float* stats = ws;
  float* feat  = ws + 1024;
  float* bufA  = ws + 4096;
  float* bufB  = bufA + 12845056;

  k_init_stats<<<4, 256, 0, stream>>>(stats);

  // ---- stage 0 : 64ch 56x56 -> dw 28x28 -> pw 10ch ----
  k_dwconv<<<50176, 256, 0, stream>>>(x, bufA, dw0_w,
      nullptr, nullptr, nullptr, 0.0f, stats + 0 * 128,
      64, 56, 56, 28, 28, 12845056);
  k_pw_wmma<64, 10, 2, 784><<<1568, 256, 0, stream>>>(bufA, stats + 0 * 128,
      bn0a_g, bn0a_b, 1.0f / 200704.0f, pw0_w, bufB, stats + 1 * 128);

  // ---- stage 1 : 10ch 28x28 -> 14x14 ----
  k_dwconv<<<1960, 256, 0, stream>>>(bufB, bufA, dwr_w + 0 * 90,
      bn0b_g, bn0b_b, stats + 1 * 128, 1.0f / 200704.0f, stats + 2 * 128,
      10, 28, 28, 14, 14, 501760);
  k_pw_wmma<10, 10, 1, 196><<<392, 256, 0, stream>>>(bufA, stats + 2 * 128,
      bnra_g + 0, bnra_b + 0, 1.0f / 50176.0f, pwr_w + 0 * 100, bufB,
      stats + 3 * 128);

  // ---- stage 2 : 14x14 -> 7x7 ----
  k_dwconv<<<490, 256, 0, stream>>>(bufB, bufA, dwr_w + 1 * 90,
      bnrb_g + 0, bnrb_b + 0, stats + 3 * 128, 1.0f / 50176.0f, stats + 4 * 128,
      10, 14, 14, 7, 7, 125440);
  k_pw_wmma<10, 10, 1, 49><<<98, 256, 0, stream>>>(bufA, stats + 4 * 128,
      bnra_g + 10, bnra_b + 10, 1.0f / 12544.0f, pwr_w + 1 * 100, bufB,
      stats + 5 * 128);

  // ---- stage 3 : 7x7 -> 4x4 ----
  k_dwconv<<<160, 256, 0, stream>>>(bufB, bufA, dwr_w + 2 * 90,
      bnrb_g + 10, bnrb_b + 10, stats + 5 * 128, 1.0f / 12544.0f, stats + 6 * 128,
      10, 7, 7, 4, 4, 40960);
  k_pw_wmma<10, 10, 1, 16><<<32, 256, 0, stream>>>(bufA, stats + 6 * 128,
      bnra_g + 20, bnra_b + 20, 1.0f / 4096.0f, pwr_w + 2 * 100, bufB,
      stats + 7 * 128);

  // ---- pool + LSTM scan ----
  k_pool<<<10, 256, 0, stream>>>(bufB, stats + 7 * 128, bnrb_g + 20, bnrb_b + 20,
      1.0f / 4096.0f, feat);
  k_lstm<<<1, 256, 0, stream>>>(feat, wih0, whh0, bih0, bhh0,
      wih1, whh1, bih1, bhh1, head_w, head_b, outp);
}
